// MyBertAttnBPWordPiece_73993696575890
// MI455X (gfx1250) — compile-verified, matrix operand order checked by reading, not run on hardware
//
#include <hip/hip_runtime.h>

// ---------------------------------------------------------------------------
// Problem constants (from reference)
// ---------------------------------------------------------------------------
#define BB      64
#define SS      512
#define WW      128
#define HH      768
#define KSUB    5          // MAX_SUB + 1
#define INTENT  26
#define SLOT    121
#define SLOTP   128        // SLOT padded to tile multiple

typedef unsigned short bf16_t;

typedef __attribute__((ext_vector_type(16))) __bf16 bf16x16;
typedef __attribute__((ext_vector_type(8)))  __bf16 bf16x8;
typedef __attribute__((ext_vector_type(8)))  float  floatx8;

__device__ __forceinline__ bf16_t f2bf(float f) {
    unsigned int u = __float_as_uint(f);
    u += 0x7FFFu + ((u >> 16) & 1u);       // round-to-nearest-even
    return (bf16_t)(u >> 16);
}

// ---------------------------------------------------------------------------
// Weight convert + transpose: W[K,N] f32 -> WT[NT,K] bf16 (rows >= N zeroed)
// ---------------------------------------------------------------------------
__global__ void wconv_kernel(const float* __restrict__ W, bf16_t* __restrict__ WT,
                             int Kdim, int N, int NT) {
    int n = blockIdx.x;                    // 0..NT-1
    for (int k = threadIdx.x; k < Kdim; k += blockDim.x) {
        float v = (n < N) ? W[(size_t)k * N + n] : 0.f;
        WT[(size_t)n * Kdim + k] = f2bf(v);
    }
}

// ---------------------------------------------------------------------------
// Gather subword segments; emit bf16 seg rows + f32/bf16 seg0 rows
// ---------------------------------------------------------------------------
__global__ void gather_kernel(const float* __restrict__ res_all,
                              const int* __restrict__ starts,
                              bf16_t* __restrict__ seg_bf,
                              bf16_t* __restrict__ seg0_bf,
                              float*  __restrict__ seg0_f) {
    int row = blockIdx.x;                  // (b*W + w)*KSUB + k
    int k   = row % KSUB;
    int bw  = row / KSUB;
    int b   = bw / WW, w = bw % WW;
    int idx = starts[b * WW + w] + k;
    idx = idx < 0 ? 0 : (idx > SS - 1 ? SS - 1 : idx);
    const float* src = res_all + ((size_t)b * SS + idx) * HH;
    for (int h = threadIdx.x; h < HH; h += blockDim.x) {
        float v = src[h];
        seg_bf[(size_t)row * HH + h] = f2bf(v);
        if (k == 0) {
            seg0_bf[(size_t)bw * HH + h] = f2bf(v);
            seg0_f [(size_t)bw * HH + h] = v;
        }
    }
}

// ---------------------------------------------------------------------------
// WMMA GEMM: C[M,ldc] = (relu?)(A[M,K]bf16 @ BT[N,K]bf16^T + bias[N])
// Macro-tile: each wave computes 32(M) x 64(N) = 2x4 WMMA 16x16 tiles,
// stepping K by 32 with v_wmma_f32_16x16x32_bf16. A fragments reused across
// 4 N-tiles, B fragments across 2 M-tiles: 12 B128 loads / 8 WMMAs per step.
// Fragment layout (16-bit A 16x32): lane row/col = L&15; K-half = (L>>4)*8;
// vector elems 0..7 -> K = khalf+0..7, elems 8..15 -> K = 16+khalf+0..7.
// BT rows are columns of B, loaded identically (K-major per lane).
// ---------------------------------------------------------------------------
__global__ void __launch_bounds__(256)
gemm_bf16_wmma_kernel(const bf16_t* __restrict__ A, const bf16_t* __restrict__ BT,
                      const float* __restrict__ bias, float* __restrict__ C,
                      int M, int N, int Kdim, int ldc, int nValid, int doRelu) {
    int lane   = threadIdx.x & 31;
    int wave   = threadIdx.x >> 5;
    int tilesN = N >> 6;                            // 64-wide N macro-tiles
    int tile   = blockIdx.x * (blockDim.x >> 5) + wave;
    if (tile >= (M >> 5) * tilesN) return;          // wave-uniform: EXEC all-ones
    int tm = (tile / tilesN) << 5;                  // 32 rows
    int tn = (tile % tilesN) << 6;                  // 64 cols
    int l15   = lane & 15;
    int khalf = (lane >> 4) << 3;                   // 0 or 8
    const bf16_t* arow0 = A + (size_t)(tm + l15) * Kdim;
    const bf16_t* arow1 = arow0 + (size_t)16 * Kdim;
    const bf16_t* brow0 = BT + (size_t)(tn + l15) * Kdim;

    floatx8 acc[2][4];
#pragma unroll
    for (int i = 0; i < 2; ++i)
#pragma unroll
        for (int j = 0; j < 4; ++j)
            acc[i][j] = (floatx8){0.f, 0.f, 0.f, 0.f, 0.f, 0.f, 0.f, 0.f};

    for (int k0 = 0; k0 < Kdim; k0 += 32) {
        bf16x8 a0lo = *(const bf16x8*)(arow0 + k0 + khalf);
        bf16x8 a0hi = *(const bf16x8*)(arow0 + k0 + 16 + khalf);
        bf16x8 a1lo = *(const bf16x8*)(arow1 + k0 + khalf);
        bf16x8 a1hi = *(const bf16x8*)(arow1 + k0 + 16 + khalf);
        bf16x16 a0 = __builtin_shufflevector(a0lo, a0hi, 0,1,2,3,4,5,6,7,8,9,10,11,12,13,14,15);
        bf16x16 a1 = __builtin_shufflevector(a1lo, a1hi, 0,1,2,3,4,5,6,7,8,9,10,11,12,13,14,15);
#pragma unroll
        for (int j = 0; j < 4; ++j) {
            const bf16_t* brow = brow0 + (size_t)(j * 16) * Kdim;
            bf16x8 blo = *(const bf16x8*)(brow + k0 + khalf);
            bf16x8 bhi = *(const bf16x8*)(brow + k0 + 16 + khalf);
            bf16x16 bv = __builtin_shufflevector(blo, bhi, 0,1,2,3,4,5,6,7,8,9,10,11,12,13,14,15);
            acc[0][j] = __builtin_amdgcn_wmma_f32_16x16x32_bf16(
                            false, a0, false, bv, (short)0, acc[0][j], false, false);
            acc[1][j] = __builtin_amdgcn_wmma_f32_16x16x32_bf16(
                            false, a1, false, bv, (short)0, acc[1][j], false, false);
        }
    }

    // C/D layout: lane col = (L&15); VGPR r -> row (L>>4)*8 + r
    int mrow = tm + ((lane >> 4) << 3);
#pragma unroll
    for (int i = 0; i < 2; ++i) {
#pragma unroll
        for (int j = 0; j < 4; ++j) {
            int col = tn + j * 16 + l15;
            if (col < nValid) {
                float bv = bias[col];
#pragma unroll
                for (int r = 0; r < 8; ++r) {
                    float v = acc[i][j][r] + bv;
                    if (doRelu) v = fmaxf(v, 0.f);
                    C[(size_t)(mrow + i * 16 + r) * ldc + col] = v;
                }
            }
        }
    }
}

// ---------------------------------------------------------------------------
// Stage-1 attention over K=5 subwords (one block per (b,w))
// ---------------------------------------------------------------------------
__global__ void attn1_kernel(const float* __restrict__ Q0, const float* __restrict__ Kk,
                             const float* __restrict__ Vv, const float* __restrict__ seg0,
                             const int* __restrict__ sublen, float* __restrict__ word) {
    int row = blockIdx.x;                  // b*W + w
    int t   = threadIdx.x;                 // 0..127
    __shared__ float red[128];
    __shared__ float sc[KSUB];
    __shared__ float at[KSUB];
    int len = sublen[row];
    const float* q = Q0 + (size_t)row * HH;
    for (int k = 0; k < KSUB; ++k) {
        const float* kr = Kk + ((size_t)row * KSUB + k) * HH;
        float p = 0.f;
        for (int h = t; h < HH; h += 128) p += q[h] * kr[h];
        red[t] = p; __syncthreads();
        for (int s = 64; s > 0; s >>= 1) { if (t < s) red[t] += red[t + s]; __syncthreads(); }
        if (t == 0) sc[k] = red[0];
        __syncthreads();
    }
    if (t == 0) {
        float m = -1e30f;
        for (int k = 0; k < KSUB; ++k) { float v = (k <= len) ? sc[k] : -1e30f; sc[k] = v; m = fmaxf(m, v); }
        float s = 0.f;
        for (int k = 0; k < KSUB; ++k) { float e = (k <= len) ? expf(sc[k] - m) : 0.f; at[k] = e; s += e; }
        for (int k = 0; k < KSUB; ++k) at[k] /= s;
    }
    __syncthreads();
    float a0 = at[0], a1 = at[1], a2 = at[2], a3 = at[3], a4 = at[4];
    const float* s0 = seg0 + (size_t)row * HH;
    const float* vr = Vv + (size_t)row * KSUB * HH;
    for (int h = t; h < HH; h += 128) {
        float base = s0[h];
        float out  = base;
        if (len != 1)
            out = base + a0 * vr[h] + a1 * vr[HH + h] + a2 * vr[2 * HH + h]
                       + a3 * vr[3 * HH + h] + a4 * vr[4 * HH + h];
        word[(size_t)row * HH + h] = out;
    }
}

// ---------------------------------------------------------------------------
// all_emb[b,w] = (w < W-1) ? word[b,w+1] : 0   (bf16 for the Wint GEMM)
// ---------------------------------------------------------------------------
__global__ void build_allemb_kernel(const float* __restrict__ word, bf16_t* __restrict__ allemb) {
    int row = blockIdx.x;                  // b*W + w
    int b = row / WW, w = row % WW;
    for (int h = threadIdx.x; h < HH; h += blockDim.x) {
        float v = (w < WW - 1) ? word[((size_t)b * WW + w + 1) * HH + h] : 0.f;
        allemb[(size_t)row * HH + h] = f2bf(v);
    }
}

// ---------------------------------------------------------------------------
// Stage-2 attention over W=128 words (one block per batch; thread t == w)
// ---------------------------------------------------------------------------
__global__ void attn2_kernel(const float* __restrict__ res, const float* __restrict__ Wh,
                             const float* __restrict__ word, float* __restrict__ h_int) {
    int b = blockIdx.x;
    int t = threadIdx.x;                   // 0..127 == w
    __shared__ float s2[128];
    __shared__ float red[128];
    const float* cls = res + (size_t)b * HH;
    const float* wh  = Wh + ((size_t)b * WW + t) * HH;
    float p = 0.f;
    for (int h = 0; h < HH; ++h) p += cls[h] * wh[h];
    s2[t] = p / sqrtf((float)HH);
    __syncthreads();
    red[t] = s2[t]; __syncthreads();
    for (int s = 64; s > 0; s >>= 1) { if (t < s) red[t] = fmaxf(red[t], red[t + s]); __syncthreads(); }
    float m = red[0]; __syncthreads();
    float e = expf(s2[t] - m);
    red[t] = e; __syncthreads();
    for (int s = 64; s > 0; s >>= 1) { if (t < s) red[t] += red[t + s]; __syncthreads(); }
    float sum = red[0]; __syncthreads();
    s2[t] = e / sum;                       // a2
    __syncthreads();
    for (int h = t; h < HH; h += 128) {
        float acc = cls[h];
        for (int w = 0; w < WW - 1; ++w)   // w = W-1 row of all_emb is zero
            acc += s2[w] * word[((size_t)b * WW + (w + 1)) * HH + h];
        h_int[(size_t)b * HH + h] = acc;
    }
}

// ---------------------------------------------------------------------------
// slot_joint[b,i] = h_intent[b] + (i < W ? word[b,i] : 0)  -> bf16
// ---------------------------------------------------------------------------
__global__ void build_sj_kernel(const float* __restrict__ h_int, const float* __restrict__ word,
                                bf16_t* __restrict__ sj) {
    int row = blockIdx.x;                  // b*(W+1) + i
    int b = row / (WW + 1), i = row % (WW + 1);
    for (int h = threadIdx.x; h < HH; h += blockDim.x) {
        float v = h_int[(size_t)b * HH + h] + ((i < WW) ? word[((size_t)b * WW + i) * HH + h] : 0.f);
        sj[(size_t)row * HH + h] = f2bf(v);
    }
}

// ---------------------------------------------------------------------------
// res_id[b,n] = h_intent[b] . Wid[:,n] + bid[n]  (tiny GEMV, 64x26)
// ---------------------------------------------------------------------------
__global__ void resid_kernel(const float* __restrict__ h_int, const float* __restrict__ Wid,
                             const float* __restrict__ bid, float* __restrict__ out) {
    int b = blockIdx.x;
    int n = threadIdx.x;
    if (n >= INTENT) return;
    float acc = bid[n];
    const float* hv = h_int + (size_t)b * HH;
    for (int h = 0; h < HH; ++h) acc += hv[h] * Wid[(size_t)h * INTENT + n];
    out[(size_t)b * INTENT + n] = acc;
}

// ---------------------------------------------------------------------------
// Launch
// ---------------------------------------------------------------------------
extern "C" void kernel_launch(void* const* d_in, const int* in_sizes, int n_in,
                              void* d_out, int out_size, void* d_ws, size_t ws_size,
                              hipStream_t stream) {
    (void)in_sizes; (void)n_in; (void)out_size; (void)ws_size;

    const float* res_all = (const float*)d_in[0];
    const float* res     = (const float*)d_in[1];
    const int*   starts  = (const int*)  d_in[2];
    const int*   sublen  = (const int*)  d_in[3];
    const float* Wsq  = (const float*)d_in[4];
    const float* Wsk  = (const float*)d_in[5];
    const float* Wsv  = (const float*)d_in[6];
    const float* Wint = (const float*)d_in[7];
    const float* Wid  = (const float*)d_in[8];
    const float* Wslot= (const float*)d_in[9];
    const float* bsq  = (const float*)d_in[10];
    const float* bsk  = (const float*)d_in[11];
    const float* bsv  = (const float*)d_in[12];
    const float* bint = (const float*)d_in[13];
    const float* bid  = (const float*)d_in[14];
    const float* bslot= (const float*)d_in[15];
    float* out = (float*)d_out;

    // workspace carve-out (256B aligned), ~458 MB total
    char* wp = (char*)d_ws;
    auto carve = [&](size_t bytes) { char* p = wp; wp += (bytes + 255) & ~(size_t)255; return p; };
    const size_t MW  = (size_t)BB * WW;          // 8192
    const size_t MKV = MW * KSUB;                // 40960
    const size_t MSJ = (size_t)BB * (WW + 1);    // 8256

    bf16_t* WsqT   = (bf16_t*)carve((size_t)HH * HH * 2);
    bf16_t* WskT   = (bf16_t*)carve((size_t)HH * HH * 2);
    bf16_t* WsvT   = (bf16_t*)carve((size_t)HH * HH * 2);
    bf16_t* WintT  = (bf16_t*)carve((size_t)HH * HH * 2);
    bf16_t* WslotT = (bf16_t*)carve((size_t)SLOTP * HH * 2);
    bf16_t* seg_bf   = (bf16_t*)carve(MKV * HH * 2);
    bf16_t* seg0_bf  = (bf16_t*)carve(MW * HH * 2);
    float*  seg0_f   = (float*)carve(MW * HH * 4);
    float*  Q0       = (float*)carve(MW * HH * 4);
    float*  Kk       = (float*)carve(MKV * HH * 4);
    float*  Vv       = (float*)carve(MKV * HH * 4);
    float*  word     = (float*)carve(MW * HH * 4);
    bf16_t* allembbf = (bf16_t*)carve(MW * HH * 2);
    float*  Wh       = (float*)carve(MW * HH * 4);
    float*  h_int    = (float*)carve((size_t)BB * HH * 4);
    bf16_t* sj_bf    = (bf16_t*)carve(MSJ * HH * 2);

    // 1. weight convert/transpose
    wconv_kernel<<<HH,    256, 0, stream>>>(Wsq,   WsqT,   HH, HH,   HH);
    wconv_kernel<<<HH,    256, 0, stream>>>(Wsk,   WskT,   HH, HH,   HH);
    wconv_kernel<<<HH,    256, 0, stream>>>(Wsv,   WsvT,   HH, HH,   HH);
    wconv_kernel<<<HH,    256, 0, stream>>>(Wint,  WintT,  HH, HH,   HH);
    wconv_kernel<<<SLOTP, 256, 0, stream>>>(Wslot, WslotT, HH, SLOT, SLOTP);

    // 2. gather segments
    gather_kernel<<<(int)MKV, 256, 0, stream>>>(res_all, starts, seg_bf, seg0_bf, seg0_f);

    // GEMM launch geometry: 8 waves/block, one 32x64 macro-tile per wave
    auto gemm_blocks = [](size_t M, int N) {
        size_t tiles = (M >> 5) * (size_t)(N >> 6);
        return (int)((tiles + 7) / 8);
    };

    // 3. Q0 / Kk / Vv GEMMs (bias + ReLU fused)
    gemm_bf16_wmma_kernel<<<gemm_blocks(MW,  HH), 256, 0, stream>>>(
        seg0_bf, WsqT, bsq, Q0, (int)MW,  HH, HH, HH, HH, 1);
    gemm_bf16_wmma_kernel<<<gemm_blocks(MKV, HH), 256, 0, stream>>>(
        seg_bf,  WskT, bsk, Kk, (int)MKV, HH, HH, HH, HH, 1);
    gemm_bf16_wmma_kernel<<<gemm_blocks(MKV, HH), 256, 0, stream>>>(
        seg_bf,  WsvT, bsv, Vv, (int)MKV, HH, HH, HH, HH, 1);

    // 4. subword attention -> word
    attn1_kernel<<<(int)MW, 128, 0, stream>>>(Q0, Kk, Vv, seg0_f, sublen, word);

    // 5. shifted embeddings -> Wh GEMM
    build_allemb_kernel<<<(int)MW, 256, 0, stream>>>(word, allembbf);
    gemm_bf16_wmma_kernel<<<gemm_blocks(MW, HH), 256, 0, stream>>>(
        allembbf, WintT, bint, Wh, (int)MW, HH, HH, HH, HH, 1);

    // 6. intent attention -> h_intent
    attn2_kernel<<<BB, 128, 0, stream>>>(res, Wh, word, h_int);

    // 7. slot joint + heads
    build_sj_kernel<<<(int)MSJ, 256, 0, stream>>>(h_int, word, sj_bf);
    resid_kernel<<<BB, 32, 0, stream>>>(h_int, Wid, bid, out);
    gemm_bf16_wmma_kernel<<<gemm_blocks(MSJ, SLOTP), 256, 0, stream>>>(
        sj_bf, WslotT, bslot, out + (size_t)BB * INTENT,
        (int)MSJ, SLOTP, HH, SLOT, SLOT, 0);
}
